// GPRGNN_26877905339089
// MI455X (gfx1250) — matrix operand order repaired and checked by reading, not compiled, not run on hardware
//
#include <hip/hip_runtime.h>
#include <hip/hip_bf16.h>

// ---------------------------------------------------------------------------
// GPR-GNN link prediction for MI455X (gfx1250, wave32).
// Dense GEMMs via v_wmma_f32_16x16x32_f16 (f16 operands, f32 accumulate).
// SpMM via edge-parallel global_atomic_add_f32 (h is 51MB -> L2 resident).
// ---------------------------------------------------------------------------

typedef __attribute__((ext_vector_type(16))) _Float16 v16h;
typedef __attribute__((ext_vector_type(8)))  _Float16 v8h;
typedef __attribute__((ext_vector_type(8)))  float    v8f;
typedef __attribute__((ext_vector_type(4)))  float    v4f;

#define D 128   // feature dim (both D_IN and D_H)

// ---- weight prep: f32 row-major W[k][n] -> f16 transposed Wt[n*128+k] ------
__global__ void prep_w16(const float* __restrict__ W, _Float16* __restrict__ Wt) {
    int j = blockIdx.x * blockDim.x + threadIdx.x;
    if (j < D * D) {
        int n = j >> 7, k = j & 127;
        Wt[j] = (_Float16)W[k * D + n];
    }
}

// ---- generic float4 zero ---------------------------------------------------
__global__ void zero_f(float* __restrict__ p, int n) {
    int i4 = (blockIdx.x * blockDim.x + threadIdx.x) * 4;
    if (i4 < n) {
        v4f z = {};
        *(v4f*)(p + i4) = z;
    }
}

// ---- degree counting + rsqrt normalization --------------------------------
__global__ void deg_count(const int* __restrict__ src, const int* __restrict__ dst,
                          float* __restrict__ dout, float* __restrict__ din, int E) {
    int e = blockIdx.x * blockDim.x + threadIdx.x;
    if (e < E) {
        atomicAdd(&dout[src[e]], 1.0f);
        atomicAdd(&din[dst[e]], 1.0f);
    }
}

__global__ void deg_rsqrt(float* __restrict__ d, int n) {
    int i = blockIdx.x * blockDim.x + threadIdx.x;
    if (i < n) {
        float v = d[i];
        d[i] = rsqrtf(v < 1.0f ? 1.0f : v);
    }
}

// ---- WMMA GEMM: out[rows,128] = act(A[rows,128] @ W[128,128] + bias) -------
// Wt is f16 with Wt[n*128+k] = W[k][n] so B-fragments are contiguous per lane.
// One block = 16 output rows; 8 waves each own one 16x16 tile (col0 = wave*16).
template<int RELU>
__global__ __launch_bounds__(256) void gemm_nk128(
    const float*    __restrict__ A,
    const _Float16* __restrict__ Wt,
    const float*    __restrict__ bias,
    float*          __restrict__ out,
    int rows)
{
    __shared__ _Float16 As[16 * D];
    const int tid  = threadIdx.x;
    const int row0 = blockIdx.x * 16;

    // stage + convert 16x128 A tile to f16 (coalesced global reads)
    #pragma unroll
    for (int i = tid; i < 16 * D; i += 256)
        As[i] = (_Float16)A[(size_t)row0 * D + i];
    __syncthreads();

    const int wave = tid >> 5;
    const int lane = tid & 31;
    const int nl   = lane & 15;   // row index (A) / column index (B,C)
    const int hi   = lane >> 4;   // half-wave select
    const int ncol = wave * 16 + nl;

    v8f c = {};
    #pragma unroll
    for (int kc = 0; kc < 4; ++kc) {
        // A fragment (16x32 f16): lane row = nl; halves 0-7 -> K0..K0+7,
        // halves 8-15 -> K0+16..K0+23, K0 = kc*32 + hi*8  (ISA 7.12.2)
        union { v16h v; v8h h[2]; } a, b;
        const int aK0 = kc * 32 + hi * 8;
        a.h[0] = *(const v8h*)&As[nl * D + aK0];
        a.h[1] = *(const v8h*)&As[nl * D + aK0 + 16];
        // B fragment (32x16 f16): lane col = nl; same K order as A:
        // halves 0-7 -> K0..K0+7, halves 8-15 -> K0+16..K0+23, K0 = kc*32 + hi*8
        const _Float16* bp = &Wt[ncol * D + kc * 32 + hi * 8];
        b.h[0] = ((const v8h*)bp)[0];
        b.h[1] = ((const v8h*)bp)[2];
        c = __builtin_amdgcn_wmma_f32_16x16x32_f16(false, a.v, false, b.v,
                                                   (short)0, c, false, false);
    }

    // C/D layout: VGPR v -> M = v + hi*8, N = nl (ISA 7.12.2)
    const float bn = bias[ncol];
    #pragma unroll
    for (int v = 0; v < 8; ++v) {
        const int m = v + hi * 8;
        float r = c[v] + bn;
        if (RELU) r = fmaxf(r, 0.0f);
        out[(size_t)(row0 + m) * D + ncol] = r;
    }
}

// ---- rep = gamma[0] * h ----------------------------------------------------
__global__ void rep_init(const float* __restrict__ h, const float* __restrict__ gamma,
                         float* __restrict__ rep, int n) {
    int i4 = (blockIdx.x * blockDim.x + threadIdx.x) * 4;
    if (i4 < n) {
        float g = gamma[0];
        v4f hv = *(const v4f*)(h + i4);
        v4f r;
        r[0] = g * hv[0]; r[1] = g * hv[1]; r[2] = g * hv[2]; r[3] = g * hv[3];
        *(v4f*)(rep + i4) = r;
    }
}

// ---- SpMM scatter: agg[dst] += h[src] * rsq_out[src] ----------------------
// 32 threads per edge, float4 per thread (128 dims). Atomics land in L2.
__global__ __launch_bounds__(256) void spmm_scatter(
    const float* __restrict__ h, const float* __restrict__ rsq_out,
    const int* __restrict__ src, const int* __restrict__ dst,
    float* __restrict__ agg, int E)
{
    int gid = blockIdx.x * 256 + threadIdx.x;
    int e = gid >> 5;
    if (e >= E) return;
    int q = (gid & 31) * 4;
    int s = src[e], d = dst[e];
    float sc = rsq_out[s];
    v4f v = *(const v4f*)(h + (size_t)s * D + q);
    float* ap = agg + (size_t)d * D + q;
    atomicAdd(ap + 0, v[0] * sc);
    atomicAdd(ap + 1, v[1] * sc);
    atomicAdd(ap + 2, v[2] * sc);
    atomicAdd(ap + 3, v[3] * sc);
}

// ---- h = agg * rsq_in;  rep += gamma[gi] * h ------------------------------
__global__ void layer_finalize(const float* __restrict__ agg, const float* __restrict__ rsq_in,
                               const float* __restrict__ gamma, int gi,
                               float* __restrict__ h, float* __restrict__ rep, int Nn)
{
    int gid = blockIdx.x * blockDim.x + threadIdx.x;
    int node = gid >> 5;
    if (node >= Nn) return;
    int q = (gid & 31) * 4;
    float r = rsq_in[node];
    float g = gamma[gi];
    size_t base = (size_t)node * D + q;
    v4f a = *(const v4f*)(agg + base);
    v4f hv, rv = *(const v4f*)(rep + base);
    #pragma unroll
    for (int j = 0; j < 4; ++j) {
        hv[j] = a[j] * r;
        rv[j] += g * hv[j];
    }
    *(v4f*)(h + base) = hv;
    *(v4f*)(rep + base) = rv;
}

// ---- candidate-edge elementwise product -----------------------------------
__global__ void pair_prod(const float* __restrict__ rep, const int* __restrict__ si,
                          const int* __restrict__ di, float* __restrict__ out, int Pn)
{
    int gid = blockIdx.x * blockDim.x + threadIdx.x;
    int i = gid >> 5;
    if (i >= Pn) return;
    int q = (gid & 31) * 4;
    int s = si[i], d = di[i];
    v4f a = *(const v4f*)(rep + (size_t)s * D + q);
    v4f b = *(const v4f*)(rep + (size_t)d * D + q);
    v4f r;
    r[0] = a[0] * b[0]; r[1] = a[1] * b[1]; r[2] = a[2] * b[2]; r[3] = a[3] * b[3];
    *(v4f*)(out + (size_t)i * D + q) = r;
}

// ---- final 128-dim dot per row, wave32 butterfly reduce -------------------
__global__ void dot_rows(const float* __restrict__ A, const float* __restrict__ w,
                         const float* __restrict__ b3, float* __restrict__ out, int rows)
{
    int gid = blockIdx.x * blockDim.x + threadIdx.x;
    int row = gid >> 5;
    if (row >= rows) return;
    int lane = gid & 31;
    float s = 0.0f;
    #pragma unroll
    for (int k = 0; k < 4; ++k) {
        int d = lane + k * 32;
        s += A[(size_t)row * D + d] * w[d];
    }
    #pragma unroll
    for (int off = 16; off > 0; off >>= 1)
        s += __shfl_xor(s, off, 32);
    if (lane == 0) out[row] = s + b3[0];
}

// ---------------------------------------------------------------------------
extern "C" void kernel_launch(void* const* d_in, const int* in_sizes, int n_in,
                              void* d_out, int out_size, void* d_ws, size_t ws_size,
                              hipStream_t stream)
{
    const float* x     = (const float*)d_in[0];
    const float* w1    = (const float*)d_in[1];
    const float* b1    = (const float*)d_in[2];
    const float* w2    = (const float*)d_in[3];
    const float* b2    = (const float*)d_in[4];
    const float* gamma = (const float*)d_in[5];
    const float* p1w   = (const float*)d_in[6];
    const float* p1b   = (const float*)d_in[7];
    const float* p2w   = (const float*)d_in[8];
    const float* p2b   = (const float*)d_in[9];
    const float* p3w   = (const float*)d_in[10];
    const float* p3b   = (const float*)d_in[11];
    const int* src     = (const int*)d_in[12];
    const int* dst     = (const int*)d_in[13];
    const int* pos_src = (const int*)d_in[14];
    const int* pos_dst = (const int*)d_in[15];
    const int* neg_src = (const int*)d_in[16];
    const int* neg_dst = (const int*)d_in[17];
    float* out = (float*)d_out;

    const int N = in_sizes[0] / D;
    const int E = in_sizes[12];
    const int P = in_sizes[14];

    // ---- workspace layout (256B aligned slots) ----
    char* ws = (char*)d_ws;
    size_t off = 0;
    auto take = [&](size_t bytes) -> char* {
        char* p = ws + off;
        off = (off + bytes + 255) & ~(size_t)255;
        return p;
    };
    _Float16* w1t  = (_Float16*)take((size_t)D * D * 2);
    _Float16* w2t  = (_Float16*)take((size_t)D * D * 2);
    _Float16* p1t  = (_Float16*)take((size_t)D * D * 2);
    _Float16* p2t  = (_Float16*)take((size_t)D * D * 2);
    float* rsq_out = (float*)take((size_t)N * 4);
    float* rsq_in  = (float*)take((size_t)N * 4);
    float* bufH    = (float*)take((size_t)N * D * 4);  // h / eprod / e2
    float* bufA    = (float*)take((size_t)N * D * 4);  // t / agg / e1
    float* rep     = (float*)take((size_t)N * D * 4);

    const int T256 = 256;
    const int gW   = (D * D + T256 - 1) / T256;      // weight prep
    const int gN   = (N + T256 - 1) / T256;          // per-node
    const int gE   = (E + T256 - 1) / T256;          // per-edge
    const int gNF4 = (N * 32 + T256 - 1) / T256;     // node x 32 lanes (f4)
    const int gEF4 = ((size_t)E * 32 + T256 - 1) / T256;
    const int gPF4 = (P * 32 + T256 - 1) / T256;
    const int gGemmN = N / 16;
    const int gGemmP = P / 16;

    // 1) weights -> f16 transposed fragments
    prep_w16<<<gW, T256, 0, stream>>>(w1, w1t);
    prep_w16<<<gW, T256, 0, stream>>>(w2, w2t);
    prep_w16<<<gW, T256, 0, stream>>>(p1w, p1t);
    prep_w16<<<gW, T256, 0, stream>>>(p2w, p2t);

    // 2) degree normalization factors
    zero_f<<<(N / 4 + T256 - 1) / T256, T256, 0, stream>>>(rsq_out, N);
    zero_f<<<(N / 4 + T256 - 1) / T256, T256, 0, stream>>>(rsq_in, N);
    deg_count<<<gE, T256, 0, stream>>>(src, dst, rsq_out, rsq_in, E);
    deg_rsqrt<<<gN, T256, 0, stream>>>(rsq_out, N);
    deg_rsqrt<<<gN, T256, 0, stream>>>(rsq_in, N);

    // 3) feature MLP: t = relu(x@W1+b1); h = t@W2+b2
    gemm_nk128<1><<<gGemmN, T256, 0, stream>>>(x, w1t, b1, bufA, N);
    gemm_nk128<0><<<gGemmN, T256, 0, stream>>>(bufA, w2t, b2, bufH, N);

    // 4) rep = gamma[0]*h; 3 propagation layers
    rep_init<<<gNF4, T256, 0, stream>>>(bufH, gamma, rep, N * D);
    for (int l = 0; l < 3; ++l) {
        zero_f<<<(N * D / 4 + T256 - 1) / T256, T256, 0, stream>>>(bufA, N * D);
        spmm_scatter<<<gEF4, T256, 0, stream>>>(bufH, rsq_out, src, dst, bufA, E);
        layer_finalize<<<gNF4, T256, 0, stream>>>(bufA, rsq_in, gamma, l + 1,
                                                  bufH, rep, N);
    }

    // 5) predictor on pos then neg candidate edges
    pair_prod<<<gPF4, T256, 0, stream>>>(rep, pos_src, pos_dst, bufH, P);
    gemm_nk128<1><<<gGemmP, T256, 0, stream>>>(bufH, p1t, p1b, bufA, P);
    gemm_nk128<1><<<gGemmP, T256, 0, stream>>>(bufA, p2t, p2b, bufH, P);
    dot_rows<<<gPF4, T256, 0, stream>>>(bufH, p3w, p3b, out, P);

    pair_prod<<<gPF4, T256, 0, stream>>>(rep, neg_src, neg_dst, bufH, P);
    gemm_nk128<1><<<gGemmP, T256, 0, stream>>>(bufH, p1t, p1b, bufA, P);
    gemm_nk128<1><<<gGemmP, T256, 0, stream>>>(bufA, p2t, p2b, bufH, P);
    dot_rows<<<gPF4, T256, 0, stream>>>(bufH, p3w, p3b, out + P, P);
}
